// LstmDiscriminator_67439576481927
// MI455X (gfx1250) — matrix operand order repaired and checked by reading, not compiled, not run on hardware
//
#include <hip/hip_runtime.h>

typedef __attribute__((ext_vector_type(16))) _Float16 v16h;
typedef __attribute__((ext_vector_type(8)))  _Float16 v8h;
typedef __attribute__((ext_vector_type(8)))  float    v8f;

__device__ __forceinline__ float leaky03(float v) { return v >= 0.f ? v : 0.3f * v; }
__device__ __forceinline__ float hsig(float v)    { return fminf(fmaxf(0.2f * v + 0.5f, 0.f), 1.f); }

// ---------------------------------------------------------------------------
// Weight prep: f32 HWIO (flat [K][COUT]) -> f16, PRE-SWIZZLED into this file's
// per-lane WMMA B-fragment layout (contiguous k, as A uses): out[chunk][ntile][lane][hi], where
//   k = chunk*32 + (lane>>4)*16 + hi,  n = ntile*16 + (lane&15)
// so each lane of a wave reads its v16h B fragment as 32 contiguous bytes.
// ---------------------------------------------------------------------------
__global__ __launch_bounds__(256)
void wswz_kernel(const float* __restrict__ w, _Float16* __restrict__ out,
                 int ktot, int cout, int ntiles, int total) {
  int gid = blockIdx.x * 256 + threadIdx.x;
  if (gid >= total) return;
  int hi    = gid & 15;
  int lane  = (gid >> 4) & 31;
  int rest  = gid >> 9;
  int ntile = rest % ntiles;
  int chunk = rest / ntiles;
  int k = chunk * 32 + ((lane >> 4) << 4) + hi;
  int n = (ntile << 4) + (lane & 15);
  out[gid] = (k < ktot) ? (_Float16)w[k * cout + n] : (_Float16)0.f;
}

// ---------------------------------------------------------------------------
// ConvLSTM2D single timestep (Keras gates i,f,c,o; hard_sigmoid; tanh)
// ---------------------------------------------------------------------------
__global__ __launch_bounds__(256)
void lstm_step_kernel(const float* __restrict__ inp, const float* __restrict__ wx,
                      const float* __restrict__ wh, const float* __restrict__ bl,
                      const float* __restrict__ h_in, const float* __restrict__ c_in,
                      float* __restrict__ h_out, float* __restrict__ c_out, int tstep) {
  __shared__ float swx[216];
  __shared__ float swh[36];
  __shared__ float sb[4];
  const int t = threadIdx.x;
  if (t < 216) swx[t] = wx[t];
  if (t < 36)  swh[t] = wh[t];
  if (t < 4)   sb[t]  = bl[t];
  __syncthreads();

  const int NPIX = 82 * 67;
  int gid = blockIdx.x * 256 + t;
  if (gid >= 64 * NPIX) return;
  int b = gid / NPIX;
  int p = gid - b * NPIX;
  int y = p / 67;
  int x = p - y * 67;

  float z0 = sb[0], z1 = sb[1], z2 = sb[2], z3 = sb[3];
  const float* xbase = inp + (size_t)(b * 24 + tstep) * NPIX * 6;
  const float* hbase = h_in + b * NPIX;

  #pragma unroll
  for (int kh = 0; kh < 3; ++kh) {
    int ih = y - 1 + kh;
    if (ih < 0 || ih >= 82) continue;
    #pragma unroll
    for (int kw = 0; kw < 3; ++kw) {
      int iw = x - 1 + kw;
      if (iw < 0 || iw >= 67) continue;
      const float* xp  = xbase + (ih * 67 + iw) * 6;
      const float* wxp = swx + (kh * 3 + kw) * 24;
      #pragma unroll
      for (int ci = 0; ci < 6; ++ci) {
        float xv = xp[ci];
        z0 += xv * wxp[ci * 4 + 0];
        z1 += xv * wxp[ci * 4 + 1];
        z2 += xv * wxp[ci * 4 + 2];
        z3 += xv * wxp[ci * 4 + 3];
      }
      float hv = hbase[ih * 67 + iw];
      const float* whp = swh + (kh * 3 + kw) * 4;
      z0 += hv * whp[0]; z1 += hv * whp[1]; z2 += hv * whp[2]; z3 += hv * whp[3];
    }
  }
  float ig = hsig(z0), fg = hsig(z1), og = hsig(z3);
  float cn = fg * c_in[gid] + ig * tanhf(z2);
  c_out[gid] = cn;
  h_out[gid] = og * tanhf(cn);
}

// ---------------------------------------------------------------------------
// concat [h | tar] -> x0 [64,82,67,6]
// ---------------------------------------------------------------------------
__global__ __launch_bounds__(256)
void concat_kernel(const float* __restrict__ h, const float* __restrict__ tar,
                   float* __restrict__ x0) {
  int gid = blockIdx.x * 256 + threadIdx.x;
  const int total = 64 * 82 * 67 * 6;
  if (gid >= total) return;
  int c  = gid % 6;
  int bp = gid / 6;
  x0[gid] = (c == 0) ? h[bp] : tar[bp * 5 + (c - 1)];
}

// ---------------------------------------------------------------------------
// Implicit-GEMM conv via V_WMMA_F32_16X16X32_F16, wave32.
// Block = 256 thr = 8 waves; block tile 64(M) x 64(N); each wave owns
// 2 M-tiles x 1 N-tile (2 accumulators -> 16 WMMAs per 32-K chunk).
// A staging is DOUBLE-BUFFERED in LDS: while chunk c is consumed from
// ldsA[c&1], the global gather for chunk c+1 is already in flight and is
// stored to ldsA[(c+1)&1] after the WMMAs issue -> 1 barrier per chunk and
// the global-load latency hides behind the matrix pipe.
// B: no LDS at all -- weights pre-swizzled in global by wswz_kernel; each
// lane loads its v16h fragment directly (2 x global_load_b128, L2-hot).
// MODE 0: leaky(acc + bias[n]);  MODE 1: BN + leaky.
// ---------------------------------------------------------------------------
template<int IH, int IW, int OH, int OW, int CIN, int COUT,
         int KH, int KW, int STR, int PH, int PW, int MODE>
__global__ __launch_bounds__(256)
void wmma_conv_kernel(const float* __restrict__ x, const _Float16* __restrict__ wswz,
                      const float* __restrict__ p0, const float* __restrict__ p1,
                      const float* __restrict__ p2, const float* __restrict__ p3,
                      float* __restrict__ out) {
  constexpr int KTOT   = KH * KW * CIN;
  constexpr int NCHUNK = (KTOT + 31) / 32;
  constexpr int NT     = COUT / 16;            // B n-tiles per chunk row
  constexpr int BUFSZ  = 4 * 32 * 16;          // halfs per LDS buffer
  __shared__ _Float16 ldsA[2][4][32][16] __attribute__((aligned(32)));

  const int t    = threadIdx.x;
  const int lane = t & 31;
  const int wv   = t >> 5;                     // 0..7
  const int mt0  = (wv >> 2) << 1;             // {0, 2}: first of 2 M sub-tiles
  const int nt   = wv & 3;                     // 0..3
  const int m0   = blockIdx.x * 64;
  const int n0   = blockIdx.y * 64;
  const int ntG  = (n0 >> 4) + nt;             // global n-tile index

  // ---- static per-thread A staging coordinates (same every chunk) ----
  const int tile = t >> 6;                     // 0..3 : which 16-row M tile
  const int r    = (t >> 2) & 15;              // row in tile
  const int cb   = (t & 3) << 3;               // k-offset block {0,8,16,24}
  const int m    = m0 + tile * 16 + r;
  const int ow   = m % OW;
  const int mm   = m / OW;
  const int oh   = mm % OH;
  const int b    = mm / OH;
  const int ln   = r + ((cb >> 4) << 4);            // LDS lane row
  const int hi0  = cb & 8;                          // LDS half offset
  _Float16* ldst = &ldsA[0][tile][ln][hi0];         // 16B-aligned

  // branch-free row-of-8 gather for k-block [k0+cb, k0+cb+8)
  auto gatherA = [&](int k0_) -> v8h {
    v8h hv;
    int k = k0_ + cb;
    if (CIN % 8 == 0) {
      // 8 consecutive k == 8 consecutive channels: one 32B vector load
      int ci = k % CIN;
      int kk = k / CIN;
      int kw = kk % KW;
      int kh = kk / KW;
      int ih = oh * STR - PH + kh;
      int iw = ow * STR - PW + kw;
      bool valid = (k < KTOT) & (ih >= 0) & (ih < IH) & (iw >= 0) & (iw < IW);
      int idx = valid ? (((b * IH + ih) * IW + iw) * CIN + ci) : 0;
      v8f xv = *(const v8f*)(x + idx);         // unconditional, 32B aligned
      #pragma unroll
      for (int j = 0; j < 8; ++j) hv[j] = valid ? (_Float16)xv[j] : (_Float16)0.f;
    } else {
      #pragma unroll
      for (int j = 0; j < 8; ++j) {
        int kj = k + j;
        int ci = kj % CIN;
        int kk = kj / CIN;
        int kw = kk % KW;
        int kh = kk / KW;
        int ih = oh * STR - PH + kh;
        int iw = ow * STR - PW + kw;
        bool valid = (kj < KTOT) & (ih >= 0) & (ih < IH) & (iw >= 0) & (iw < IW);
        int idx = valid ? (((b * IH + ih) * IW + iw) * CIN + ci) : 0;
        float v = x[idx];                      // unconditional load
        hv[j] = valid ? (_Float16)v : (_Float16)0.f;
      }
    }
    return hv;
  };

  v8f acc0 = {};
  v8f acc1 = {};

  // prologue: stage chunk 0 into buffer 0
  {
    v8h hv0 = gatherA(0);
    *(v8h*)ldst = hv0;
  }

  for (int c = 0; c < NCHUNK; ++c) {
    __syncthreads();                           // buffer c&1 fully staged

    // kick off next chunk's gather immediately (latency hides under WMMA)
    v8h hnext;
    const bool hasNext = (c + 1 < NCHUNK);     // uniform
    if (hasNext) hnext = gatherA((c + 1) * 32);

    // B fragment straight from global (L2-hot)
    const v16h* bp = (const v16h*)wswz + (c * NT + ntG) * 32 + lane;
    v16h bv = *bp;
    if (hasNext)
      __builtin_prefetch((const _Float16*)(bp + NT * 32), 0, 0);

    const _Float16* abuf = &ldsA[c & 1][0][0][0];
    v16h av0 = *(const v16h*)(abuf + ((mt0 + 0) * 32 + lane) * 16);
    v16h av1 = *(const v16h*)(abuf + ((mt0 + 1) * 32 + lane) * 16);
    acc0 = __builtin_amdgcn_wmma_f32_16x16x32_f16(false, av0, false, bv,
                                                  (short)0, acc0, false, false);
    acc1 = __builtin_amdgcn_wmma_f32_16x16x32_f16(false, av1, false, bv,
                                                  (short)0, acc1, false, false);

    // store next chunk into the other buffer (no barrier needed here:
    // consumers of this buffer sync at the top of the next iteration)
    if (hasNext)
      *(v8h*)(ldst + ((c + 1) & 1) * BUFSZ) = hnext;
  }

  // ---- epilogue: lane holds N = lane%16; rows M = (lane/16)*8 + r ----
  int n  = n0 + nt * 16 + (lane & 15);
  int mB = m0 + mt0 * 16 + ((lane >> 4) << 3);
  float s0 = 0.f, s1 = 0.f, s2 = 0.f, s3 = 0.f;
  if (MODE == 0) {
    s0 = p0[n];
  } else {
    s2 = p2[n];
    s3 = rsqrtf(p3[n] + 1e-3f) * p0[n];
    s1 = p1[n];
  }
  #pragma unroll
  for (int rr = 0; rr < 8; ++rr) {
    float v = acc0[rr];
    v = (MODE == 0) ? (v + s0) : ((v - s2) * s3 + s1);
    out[(mB + rr) * COUT + n] = leaky03(v);
  }
  #pragma unroll
  for (int rr = 0; rr < 8; ++rr) {
    float v = acc1[rr];
    v = (MODE == 0) ? (v + s0) : ((v - s2) * s3 + s1);
    out[(mB + 16 + rr) * COUT + n] = leaky03(v);
  }
}

// ---------------------------------------------------------------------------
// conv4: pad 1 + 4x4x512 valid -> 1 channel : [64,20,16,512] -> [64,19,15,1]
// ---------------------------------------------------------------------------
__global__ __launch_bounds__(256)
void conv4_kernel(const float* __restrict__ a3, const float* __restrict__ w4,
                  const float* __restrict__ b4, float* __restrict__ out) {
  const int OH = 19, OW = 15, IH = 20, IW = 16, CI = 512;
  int gid = blockIdx.x * 256 + threadIdx.x;
  if (gid >= 64 * OH * OW) return;
  int b  = gid / (OH * OW);
  int p  = gid - b * (OH * OW);
  int oh = p / OW;
  int ow = p - oh * OW;
  float acc = 0.f;
  for (int kh = 0; kh < 4; ++kh) {
    int ih = oh - 1 + kh;
    if (ih < 0 || ih >= IH) continue;
    for (int kw = 0; kw < 4; ++kw) {
      int iw = ow - 1 + kw;
      if (iw < 0 || iw >= IW) continue;
      const float4* ap = (const float4*)(a3 + ((b * IH + ih) * IW + iw) * CI);
      const float4* wp = (const float4*)(w4 + (kh * 4 + kw) * CI);
      #pragma unroll 4
      for (int c4 = 0; c4 < CI / 4; ++c4) {
        float4 a = ap[c4], w = wp[c4];
        acc += a.x * w.x + a.y * w.y + a.z * w.z + a.w * w.w;
      }
    }
  }
  out[gid] = acc + b4[0];
}

// ---------------------------------------------------------------------------
extern "C" void kernel_launch(void* const* d_in, const int* in_sizes, int n_in,
                              void* d_out, int out_size, void* d_ws, size_t ws_size,
                              hipStream_t stream) {
  (void)in_sizes; (void)n_in; (void)out_size; (void)ws_size;
  const float* inp = (const float*)d_in[0];
  const float* tar = (const float*)d_in[1];
  const float* wx  = (const float*)d_in[2];
  const float* wh  = (const float*)d_in[3];
  const float* bl  = (const float*)d_in[4];
  const float* w1  = (const float*)d_in[5];
  const float* b1  = (const float*)d_in[6];
  const float* w2  = (const float*)d_in[7];
  const float* b2  = (const float*)d_in[8];
  const float* w3  = (const float*)d_in[9];
  const float* bng = (const float*)d_in[10];
  const float* bnb = (const float*)d_in[11];
  const float* bnm = (const float*)d_in[12];
  const float* bnv = (const float*)d_in[13];
  const float* w4  = (const float*)d_in[14];
  const float* b4  = (const float*)d_in[15];
  float* out = (float*)d_out;

  const int NPIX = 82 * 67;
  char* ws = (char*)d_ws;
  size_t off = 0;
  auto alloc = [&](size_t bytes) -> void* {
    void* p = ws + off;
    off = (off + bytes + 255) & ~(size_t)255;
    return p;
  };
  float* h0 = (float*)alloc(sizeof(float) * 64 * NPIX);
  float* h1 = (float*)alloc(sizeof(float) * 64 * NPIX);
  float* c0 = (float*)alloc(sizeof(float) * 64 * NPIX);
  float* c1 = (float*)alloc(sizeof(float) * 64 * NPIX);
  float* x0 = (float*)alloc(sizeof(float) * 64 * NPIX * 6);
  float* a1 = (float*)alloc(sizeof(float) * 64 * 41 * 34 * 64);
  float* a2 = (float*)alloc(sizeof(float) * 64 * 21 * 17 * 128);
  float* a3 = (float*)alloc(sizeof(float) * 64 * 20 * 16 * 512);
  // swizzled f16 weights: chunks * (COUT/16) * 512 halfs
  _Float16* w1h = (_Float16*)alloc(sizeof(_Float16) * 5  * 4  * 512);   // K=150
  _Float16* w2h = (_Float16*)alloc(sizeof(_Float16) * 50 * 8  * 512);   // K=1600
  _Float16* w3h = (_Float16*)alloc(sizeof(_Float16) * 64 * 32 * 512);   // K=2048

  hipMemsetAsync(h0, 0, sizeof(float) * 64 * NPIX, stream);
  hipMemsetAsync(c0, 0, sizeof(float) * 64 * NPIX, stream);

  // pre-swizzle weights into WMMA B-fragment order
  wswz_kernel<<<(5 * 4 * 512 + 255) / 256, 256, 0, stream>>>(w1, w1h, 150, 64, 4, 5 * 4 * 512);
  wswz_kernel<<<(50 * 8 * 512 + 255) / 256, 256, 0, stream>>>(w2, w2h, 1600, 128, 8, 50 * 8 * 512);
  wswz_kernel<<<(64 * 32 * 512 + 255) / 256, 256, 0, stream>>>(w3, w3h, 2048, 512, 32, 64 * 32 * 512);

  // ConvLSTM: 24 sequential steps, ping-pong; final h lands in h0 (T even)
  float* hb[2] = {h0, h1};
  float* cb[2] = {c0, c1};
  const int lstm_blocks = (64 * NPIX + 255) / 256;
  for (int t = 0; t < 24; ++t) {
    lstm_step_kernel<<<lstm_blocks, 256, 0, stream>>>(
        inp, wx, wh, bl, hb[t & 1], cb[t & 1], hb[(t + 1) & 1], cb[(t + 1) & 1], t);
  }

  concat_kernel<<<(64 * NPIX * 6 + 255) / 256, 256, 0, stream>>>(h0, tar, x0);

  // conv1: 5x5 s2 SAME (pad 1,2), 6->64 : [64,82,67,6] -> [64,41,34,64]
  wmma_conv_kernel<82, 67, 41, 34, 6, 64, 5, 5, 2, 1, 2, 0>
      <<<dim3(64 * 41 * 34 / 64, 64 / 64), 256, 0, stream>>>(
          x0, w1h, b1, nullptr, nullptr, nullptr, a1);

  // conv2: 5x5 s2 SAME (pad 2,1), 64->128 : -> [64,21,17,128]
  wmma_conv_kernel<41, 34, 21, 17, 64, 128, 5, 5, 2, 2, 1, 0>
      <<<dim3(64 * 21 * 17 / 64, 128 / 64), 256, 0, stream>>>(
          a1, w2h, b2, nullptr, nullptr, nullptr, a2);

  // conv3: zero-pad 1 + 4x4 valid, 128->512, BN(eps 1e-3)+leaky fused
  wmma_conv_kernel<21, 17, 20, 16, 128, 512, 4, 4, 1, 1, 1, 1>
      <<<dim3(64 * 20 * 16 / 64, 512 / 64), 256, 0, stream>>>(
          a2, w3h, bng, bnb, bnm, bnv, a3);

  // conv4: pad 1 + 4x4 valid, 512->1, + bias : -> [64,19,15,1]
  conv4_kernel<<<(64 * 19 * 15 + 255) / 256, 256, 0, stream>>>(a3, w4, b4, out);
}
